// DataManifolds_18915035972342
// MI455X (gfx1250) — matrix operand order repaired and checked by reading, not compiled, not run on hardware
//
#include <hip/hip_runtime.h>

typedef __attribute__((ext_vector_type(16))) _Float16     v16h;
typedef __attribute__((ext_vector_type(4)))  _Float16     v4h;
typedef __attribute__((ext_vector_type(8)))  float        v8f;
typedef __attribute__((ext_vector_type(8)))  unsigned int v8u;

#define DEVINL __device__ __forceinline__

// ---------------- problem constants ----------------
constexpr int RUNS = 1000;
constexpr int WAYS = 5;
constexpr int NS   = 25;     // labeled (support) rows
constexpr int QS   = 75;     // query rows
constexpr int NTOT = 100;    // NS + QS
constexpr int DIM  = 640;
constexpr int LINKK = 20;
constexpr int EPOCHS = 5;
constexpr float LAM = 10.0f;
constexpr float ALPHA = 0.7f;
constexpr int BLK = 256;     // 8 waves (wave32)
constexpr int OT_IT = 40;    // Sinkhorn iterations (ref runs to 1e-3 tol)
constexpr int SOLVE_IT = 24; // Neumann iters: 0.7^24 ~ 2e-4 ~ f16 state resolution

// ---------------- LDS layout (bytes), total 163748 <= 160KiB ----------------
constexpr int OFF_FEAT  = 0;       // f16 featH[100*640] (raw F, then graph-conv'd in place)
constexpr int OFF_GW    = 128000;  // f16 [100*100]: A -> G -> A2 -> W
constexpr int OFF_NORM  = 148000;  // f32[100] row norms
constexpr int OFF_RS    = 148400;  // f32[100] rowsum^-1/2
constexpr int OFF_PROTO = 148800;  // f16[5*640]
constexpr int OFF_PNORM = 155200;  // f32[5] (+4 pad)
constexpr int OFF_Z     = 155224;  // f32[100*5]
constexpr int OFF_XA    = 157224;  // f16[16][100] solver state (ping)
constexpr int OFF_XB    = 160424;  // f16[16][100] solver state (pong)
constexpr int OFF_COL   = 163624;  // f32[5]
constexpr int OFF_YS    = 163644;  // int[25]
constexpr int OFF_ACC   = 163744;  // f32[1]
constexpr int LDS_BYTES = 163748;

DEVINL v8f wmma16(v16h a, v16h b, v8f c) {
  // D = A(16x32 f16) x B(32x16 f16) + C(16x16 f32)
  return __builtin_amdgcn_wmma_f32_16x16x32_f16(false, a, false, b, (short)0, c, false, false);
}

// ---- fragment loaders (contiguous k order, k = 16*(lane/16) + s, in A and B alike) ----
// Invalid M/N lanes: address is clamped to row 0 only (1 select); the garbage outputs
// land in discarded D elements. K-dimension checks are kept where K is padded (they
// constant-fold when k0 is a literal) since bad K contaminates valid outputs.

// A (16 x 32 f16): lanes 0-15 row M=lane hold K=0..15 (pairs 2j,2j+1);
// lanes 16-31 row M=lane-16 hold K=16..31.  No k-checks.
DEVINL v16h fragA_u(const _Float16* base, int stride, int m0, int k0, int mMax) {
  const int lane = threadIdx.x & 31;
  int m = m0 + (lane & 15);
  if (m >= mMax) m = 0;
  const int kb = k0 + ((lane < 16) ? 0 : 16);
  v8u u;
#pragma unroll
  for (int j = 0; j < 8; ++j) {
    const int kk = kb + 2 * j;
    u[j] = *reinterpret_cast<const unsigned*>(base + m * stride + kk);
  }
  return __builtin_bit_cast(v16h, u);
}

// A with k-checks (use where k0 is a compile-time constant so checks fold).
DEVINL v16h fragA_c(const _Float16* base, int stride, int m0, int k0, int mMax, int kMax) {
  const int lane = threadIdx.x & 31;
  int m = m0 + (lane & 15);
  if (m >= mMax) m = 0;
  const int kb = k0 + ((lane < 16) ? 0 : 16);
  v8u u;
#pragma unroll
  for (int j = 0; j < 8; ++j) {
    const int kk = kb + 2 * j;
    u[j] = (kk < kMax) ? *reinterpret_cast<const unsigned*>(base + m * stride + kk) : 0u;
  }
  return __builtin_bit_cast(v16h, u);
}

// B (32 x 16 f16) with B[k][n] = src[n][k] (transposed source): lanes 0-15 col N=lane
// hold K=0..15 (pairs 2j,2j+1); lanes 16-31 hold K=16..31.  No k-checks.
DEVINL v16h fragBT_u(const _Float16* base, int stride, int n0, int k0, int nMax) {
  const int lane = threadIdx.x & 31;
  int n = n0 + (lane & 15);
  if (n >= nMax) n = 0;
  const int kb = k0 + ((lane < 16) ? 0 : 16);
  v8u u;
#pragma unroll
  for (int j = 0; j < 8; ++j)
    u[j] = *reinterpret_cast<const unsigned*>(base + n * stride + kb + 2 * j);
  return __builtin_bit_cast(v16h, u);
}

// Transposed-source B with k-checks (kMax even -> a pair is fully valid or fully out).
DEVINL v16h fragBT_c(const _Float16* base, int stride, int n0, int k0, int nMax, int kMax) {
  const int lane = threadIdx.x & 31;
  int n = n0 + (lane & 15);
  if (n >= nMax) n = 0;
  const int kb = k0 + ((lane < 16) ? 0 : 16);
  v8u u;
#pragma unroll
  for (int j = 0; j < 8; ++j) {
    const int kk = kb + 2 * j;
    u[j] = (kk < kMax) ? *reinterpret_cast<const unsigned*>(base + n * stride + kk) : 0u;
  }
  return __builtin_bit_cast(v16h, u);
}

// B (32 x 16 f16) with B[k][n] = src[k][n] (row-major source); scalar element loads.
DEVINL v16h fragB_rm(const _Float16* base, int stride, int n0, int k0, int kMax) {
  const int lane = threadIdx.x & 31;
  const int n = n0 + (lane & 15);
  const int kb = k0 + ((lane < 16) ? 0 : 16);
  v16h b;
#pragma unroll
  for (int j = 0; j < 8; ++j) {
    const int kk = kb + 2 * j;
    b[2 * j]     = (kk < kMax)     ? base[kk * stride + n]       : (_Float16)0;
    b[2 * j + 1] = (kk + 1 < kMax) ? base[(kk + 1) * stride + n] : (_Float16)0;
  }
  return b;
}

// A fragment for Z^T (new_proto): A[m][k] = Z[k*WAYS+m], f32 source with convert.
// m>=WAYS rows read harmless garbage (discarded); k-checks fold (k0 literal).
DEVINL v16h fragA_Zt(const float* Z, int k0) {
  const int lane = threadIdx.x & 31;
  const int m = lane & 15;
  const int kb = k0 + ((lane < 16) ? 0 : 16);
  v16h a;
#pragma unroll
  for (int j = 0; j < 8; ++j) {
    const int kk = kb + 2 * j;
    a[2 * j]     = (kk < NTOT)     ? (_Float16)Z[kk * WAYS + m]       : (_Float16)0;
    a[2 * j + 1] = (kk + 1 < NTOT) ? (_Float16)Z[(kk + 1) * WAYS + m] : (_Float16)0;
  }
  return a;
}

// Pairwise affinity: gw[i][j] = exp(-lam*max(nrm[i]+nrm[j]-2*(F F^T)[i][j],0)), opt. zero diag.
template <bool ZERO_DIAG>
__device__ void gramExp(const _Float16* featH, const float* nrm, _Float16* gw,
                        int wave, int lane) {
  v8f acc[7] = {};
  for (int ks = 0; ks < DIM / 32; ++ks) {
    const int k0 = ks * 32;
#pragma unroll
    for (int idx = 0; idx < 7; ++idx) {
      const int t = wave + idx * 8;            // uniform per wave
      if (t < 49) {
        const int mi = t / 7, mj = t % 7;
        v16h a = fragA_u(featH, DIM, mi * 16, k0, NTOT);
        v16h b = fragBT_u(featH, DIM, mj * 16, k0, NTOT);
        acc[idx] = wmma16(a, b, acc[idx]);
      }
    }
  }
#pragma unroll
  for (int idx = 0; idx < 7; ++idx) {
    const int t = wave + idx * 8;
    if (t < 49) {
      const int mi = t / 7, mj = t % 7;
      const int col = mj * 16 + (lane & 15);
#pragma unroll
      for (int r = 0; r < 8; ++r) {
        const int row = mi * 16 + r + ((lane < 16) ? 0 : 8);
        if (row < NTOT && col < NTOT) {
          const float d2 = fmaxf(nrm[row] + nrm[col] - 2.0f * acc[idx][r], 0.0f);
          float v = __expf(-LAM * d2);
          if (ZERO_DIAG && row == col) v = 0.0f;
          gw[row * NTOT + col] = (_Float16)v;
        }
      }
    }
  }
}

// dist(featH, proto) -> Z rows: support = one-hot(ys), query = exp(-lam*d2).
// WMMA over K=640, 4 interleaved accumulators (no dependent-WMMA NOPs).
__device__ void fillZ_wmma(const _Float16* featH, const _Float16* proto, const float* nrm,
                           const float* pnorm, const int* ysl, float* Z, int wave, int lane) {
  if (wave < 7) {
    v8f d0 = {}, d1 = {}, d2 = {}, d3 = {};
    for (int ks = 0; ks < DIM / 32; ks += 4) {
      d0 = wmma16(fragA_u(featH, DIM, wave * 16, (ks + 0) * 32, NTOT),
                  fragBT_u(proto, DIM, 0, (ks + 0) * 32, WAYS), d0);
      d1 = wmma16(fragA_u(featH, DIM, wave * 16, (ks + 1) * 32, NTOT),
                  fragBT_u(proto, DIM, 0, (ks + 1) * 32, WAYS), d1);
      d2 = wmma16(fragA_u(featH, DIM, wave * 16, (ks + 2) * 32, NTOT),
                  fragBT_u(proto, DIM, 0, (ks + 2) * 32, WAYS), d2);
      d3 = wmma16(fragA_u(featH, DIM, wave * 16, (ks + 3) * 32, NTOT),
                  fragBT_u(proto, DIM, 0, (ks + 3) * 32, WAYS), d3);
    }
    const int n = lane & 15;
#pragma unroll
    for (int r = 0; r < 8; ++r) {
      const int row = wave * 16 + r + ((lane < 16) ? 0 : 8);
      if (row < NTOT && n < WAYS) {
        float v;
        if (row < NS) {
          v = (ysl[row] == n) ? 1.0f : 0.0f;
        } else {
          const float dot = d0[r] + d1[r] + d2[r] + d3[r];
          const float d2v = fmaxf(nrm[row] + pnorm[n] - 2.0f * dot, 0.0f);
          v = __expf(-LAM * d2v);
        }
        Z[row * WAYS + n] = v;
      }
    }
  }
}

DEVINL float waveSum(float v) {
  v += __shfl_xor(v, 1);
  v += __shfl_xor(v, 2);
  v += __shfl_xor(v, 4);
  v += __shfl_xor(v, 8);
  v += __shfl_xor(v, 16);
  return v;
}

// Sinkhorn on query rows (n_l=0): single-wave, register-resident, shuffle col-sums.
__device__ void otQueryW(float* Z, int lane) {
  float p[3][WAYS];
#pragma unroll
  for (int c = 0; c < 3; ++c) {
    const int row = NS + lane + 32 * c;
    const bool va = (row < NTOT);
#pragma unroll
    for (int k = 0; k < WAYS; ++k) p[c][k] = va ? Z[row * WAYS + k] : 0.0f;
  }
  for (int it = 0; it < OT_IT; ++it) {
#pragma unroll
    for (int c = 0; c < 3; ++c) {          // row normalize (zero rows stay zero)
      float u = 0.0f;
#pragma unroll
      for (int k = 0; k < WAYS; ++k) u += p[c][k];
      const float inv = 1.0f / fmaxf(u, 1e-30f);
#pragma unroll
      for (int k = 0; k < WAYS; ++k) p[c][k] *= inv;
    }
#pragma unroll
    for (int k = 0; k < WAYS; ++k) {       // column normalize
      const float cs = waveSum(p[0][k] + p[1][k] + p[2][k]);
      const float sc = (float)(QS / WAYS) / fmaxf(cs, 1e-30f);
#pragma unroll
      for (int c = 0; c < 3; ++c) p[c][k] *= sc;
    }
  }
#pragma unroll
  for (int c = 0; c < 3; ++c) {
    const int row = NS + lane + 32 * c;
    if (row < NTOT) {
#pragma unroll
      for (int k = 0; k < WAYS; ++k) Z[row * WAYS + k] = p[c][k];
    }
  }
}

// Sinkhorn on all rows, first NS clamped to one-hot each iteration; single-wave.
__device__ void otFullW(float* Z, const int* ysl, int lane) {
  float p[4][WAYS];
#pragma unroll
  for (int c = 0; c < 4; ++c) {
    const int row = lane + 32 * c;
    const bool va = (row < NTOT);
#pragma unroll
    for (int k = 0; k < WAYS; ++k) p[c][k] = va ? Z[row * WAYS + k] : 0.0f;
  }
  const int myy = (lane < NS) ? ysl[lane] : -1;   // clamped rows live in c==0
  for (int it = 0; it < OT_IT; ++it) {
#pragma unroll
    for (int c = 0; c < 4; ++c) {
      float u = 0.0f;
#pragma unroll
      for (int k = 0; k < WAYS; ++k) u += p[c][k];
      const float inv = 1.0f / fmaxf(u, 1e-30f);
#pragma unroll
      for (int k = 0; k < WAYS; ++k) p[c][k] *= inv;
    }
#pragma unroll
    for (int k = 0; k < WAYS; ++k) {
      const float cs = waveSum(p[0][k] + p[1][k] + p[2][k] + p[3][k]);
      const float sc = (float)(NTOT / WAYS) / fmaxf(cs, 1e-30f);
#pragma unroll
      for (int c = 0; c < 4; ++c) p[c][k] *= sc;
    }
    if (myy >= 0) {
#pragma unroll
      for (int k = 0; k < WAYS; ++k) p[0][k] = (myy == k) ? 1.0f : 0.0f;
    }
  }
#pragma unroll
  for (int c = 0; c < 4; ++c) {
    const int row = lane + 32 * c;
    if (row < NTOT) {
#pragma unroll
      for (int k = 0; k < WAYS; ++k) Z[row * WAYS + k] = p[c][k];
    }
  }
}

__global__ __launch_bounds__(BLK)
void data_manifolds_kernel(const float* __restrict__ xs, const float* __restrict__ xq,
                           const int* __restrict__ ysg, const int* __restrict__ yqg,
                           float* __restrict__ out) {
  __shared__ __align__(16) unsigned char lds[LDS_BYTES];
  _Float16* featH = reinterpret_cast<_Float16*>(lds + OFF_FEAT);
  _Float16* gw    = reinterpret_cast<_Float16*>(lds + OFF_GW);
  float*    nrm   = reinterpret_cast<float*>(lds + OFF_NORM);
  float*    rs    = reinterpret_cast<float*>(lds + OFF_RS);
  _Float16* proto = reinterpret_cast<_Float16*>(lds + OFF_PROTO);
  float*    pnorm = reinterpret_cast<float*>(lds + OFF_PNORM);
  float*    Z     = reinterpret_cast<float*>(lds + OFF_Z);
  _Float16* xa    = reinterpret_cast<_Float16*>(lds + OFF_XA);
  _Float16* xb    = reinterpret_cast<_Float16*>(lds + OFF_XB);
  float*    colb  = reinterpret_cast<float*>(lds + OFF_COL);
  int*      ysl   = reinterpret_cast<int*>(lds + OFF_YS);
  float*    accv  = reinterpret_cast<float*>(lds + OFF_ACC);

  const int run  = blockIdx.x;
  const int tid  = threadIdx.x;
  const int wave = tid >> 5;
  const int lane = tid & 31;

  // ---- Prologue: labels; F (global f32, b128 loads) -> featH (f16, b64 LDS stores) ----
  if (tid < NS) ysl[tid] = ysg[run * NS + tid];
  for (int id = tid; id < (NTOT * DIM) / 4; id += BLK) {
    const int row = id / (DIM / 4);
    const int q   = id % (DIM / 4);
    const float4 v = (row < NS)
        ? reinterpret_cast<const float4*>(xs + ((size_t)run * NS + row) * DIM)[q]
        : reinterpret_cast<const float4*>(xq + ((size_t)run * QS + (row - NS)) * DIM)[q];
    v4h h;
    h[0] = (_Float16)v.x; h[1] = (_Float16)v.y; h[2] = (_Float16)v.z; h[3] = (_Float16)v.w;
    *reinterpret_cast<v4h*>(featH + row * DIM + 4 * q) = h;
  }
  __syncthreads();

  // ---- raw-feature row norms ----
  if (tid < NTOT) {
    float s = 0.0f;
    for (int d = 0; d < DIM; ++d) { const float v = (float)featH[tid * DIM + d]; s += v * v; }
    nrm[tid] = s;
  }
  __syncthreads();

  // ---- Stage 1: A = exp(-lam*sqdist(F,F)) f16 in gw (WMMA Gram, exp at writeback) ----
  gramExp<false>(featH, nrm, gw, wave, lane);
  __syncthreads();

  // ---- graph conv weights: G = (0.5 I + 0.5 * D^-1/2 A D^-1/2)^4, in place over gw ----
  if (tid < NTOT) {
    float s = 0.0f;
    for (int j = 0; j < NTOT; ++j) s += (float)gw[tid * NTOT + j];
    rs[tid] = rsqrtf(fmaxf(s, 1e-30f));
  }
  __syncthreads();
  for (int id = tid; id < NTOT * NTOT; id += BLK) {
    const int i = id / NTOT, j = id % NTOT;
    float g = 0.5f * ((i == j) ? 1.0f : 0.0f) + 0.5f * rs[i] * rs[j] * (float)gw[id];
    g = g * g; g = g * g;                      // elementwise ^4 (K_HOP = 4)
    gw[id] = (_Float16)g;
  }
  __syncthreads();

  // ---- Stage 2: featH = G @ featH, IN PLACE (disjoint col-blocks per wave) ----
  for (int nj = wave; nj < DIM / 16; nj += 8) {
    v8f c[7] = {};
#pragma unroll
    for (int ksx = 0; ksx < 4; ++ksx) {
      const int k0 = ksx * 32;
      v16h b = fragB_rm(featH, DIM, nj * 16, k0, NTOT);        // k-checks fold
#pragma unroll
      for (int mi = 0; mi < 7; ++mi) {
        v16h a = fragA_c(gw, NTOT, mi * 16, k0, NTOT, NTOT);   // k-checks fold
        c[mi] = wmma16(a, b, c[mi]);
      }
    }
    const int col = nj * 16 + (lane & 15);
#pragma unroll
    for (int mi = 0; mi < 7; ++mi) {
#pragma unroll
      for (int r = 0; r < 8; ++r) {
        const int row = mi * 16 + r + ((lane < 16) ? 0 : 8);
        if (row < NTOT) featH[row * DIM + col] = (_Float16)c[mi][r];
      }
    }
  }
  __syncthreads();

  // ---- updated norms; prototype init (mean of 5 support rows per class) ----
  if (tid < NTOT) {
    float s = 0.0f;
    for (int d = 0; d < DIM; ++d) { const float v = (float)featH[tid * DIM + d]; s += v * v; }
    nrm[tid] = s;
  }
  for (int id = tid; id < WAYS * DIM; id += BLK) {
    const int k = id / DIM, d = id % DIM;
    float s = 0.0f;
#pragma unroll
    for (int sh = 0; sh < 5; ++sh) s += (float)featH[(k * 5 + sh) * DIM + d];
    proto[id] = (_Float16)(s * 0.2f);
  }
  __syncthreads();
  if (tid < WAYS) {
    float s = 0.0f;
    for (int d = 0; d < DIM; ++d) { const float v = (float)proto[tid * DIM + d]; s += v * v; }
    pnorm[tid] = s;
  }
  __syncthreads();

  // ---- Stage 4: A2 = exp(-lam*sqdist(featH,featH)) with zero diag, into gw ----
  gramExp<true>(featH, nrm, gw, wave, lane);
  __syncthreads();

  // ---- build_graph: top-20 per row, then W = D^-1/2 W D^-1/2 (f16 in place) ----
  if (tid < NTOT) {
    _Float16* wr = gw + tid * NTOT;
    unsigned long long mA = 0ull, mB = 0ull;
    for (int t = 0; t < LINKK; ++t) {
      float best = -1.0f; int bj = 0;
      for (int j = 0; j < NTOT; ++j) {
        const bool taken = (j < 64) ? ((mA >> j) & 1ull) : ((mB >> (j - 64)) & 1ull);
        const float v = (float)wr[j];
        if (!taken && v > best) { best = v; bj = j; }
      }
      if (bj < 64) mA |= (1ull << bj); else mB |= (1ull << (bj - 64));
    }
    float s = 0.0f;
    for (int j = 0; j < NTOT; ++j) {
      const bool keep = (j < 64) ? ((mA >> j) & 1ull) : ((mB >> (j - 64)) & 1ull);
      if (!keep) wr[j] = (_Float16)0; else s += (float)wr[j];
    }
    rs[tid] = rsqrtf(fmaxf(s, 1e-30f));
  }
  __syncthreads();
  for (int id = tid; id < NTOT * NTOT; id += BLK) {
    const int i = id / NTOT, j = id % NTOT;
    gw[id] = (_Float16)((float)gw[id] * rs[i] * rs[j]);
  }
  __syncthreads();

  // ---- EPOCHS of label propagation ----
  for (int ep = 0; ep < EPOCHS; ++ep) {
    fillZ_wmma(featH, proto, nrm, pnorm, ysl, Z, wave, lane);
    __syncthreads();
    if (wave == 0) otQueryW(Z, lane);
    __syncthreads();

    // X = (I - a*W)^-1 Z via X <- Z + a*W*X (WMMA, ping-pong f16 X tiles, 1 barrier/iter)
    for (int id = tid; id < NTOT * WAYS; id += BLK) {
      const int i = id / WAYS, k = id % WAYS;
      xa[k * NTOT + i] = (_Float16)Z[id];
    }
    __syncthreads();
    for (int t = 0; t < SOLVE_IT; ++t) {
      const _Float16* xsrc = (t & 1) ? xb : xa;
      _Float16*       xdst = (t & 1) ? xa : xb;
      if (wave < 7) {
        v8f d0 = {}, d1 = {};
        d0 = wmma16(fragA_c(gw, NTOT, wave * 16,  0, NTOT, NTOT),
                    fragBT_c(xsrc, NTOT, 0,  0, 16, NTOT), d0);
        d1 = wmma16(fragA_c(gw, NTOT, wave * 16, 32, NTOT, NTOT),
                    fragBT_c(xsrc, NTOT, 0, 32, 16, NTOT), d1);
        d0 = wmma16(fragA_c(gw, NTOT, wave * 16, 64, NTOT, NTOT),
                    fragBT_c(xsrc, NTOT, 0, 64, 16, NTOT), d0);
        d1 = wmma16(fragA_c(gw, NTOT, wave * 16, 96, NTOT, NTOT),
                    fragBT_c(xsrc, NTOT, 0, 96, 16, NTOT), d1);
        const int n = lane & 15;
#pragma unroll
        for (int r = 0; r < 8; ++r) {
          const int row = wave * 16 + r + ((lane < 16) ? 0 : 8);
          if (row < NTOT && n < WAYS)
            xdst[n * NTOT + row] = (_Float16)(Z[row * WAYS + n] + ALPHA * (d0[r] + d1[r]));
        }
      }
      __syncthreads();
    }
    for (int id = tid; id < NTOT * WAYS; id += BLK) {   // SOLVE_IT even -> result in xa
      const int i = id / WAYS, k = id % WAYS;
      Z[id] = fmaxf((float)xa[k * NTOT + i], 0.0f);
    }
    __syncthreads();
    if (wave == 0) otFullW(Z, ysl, lane);
    __syncthreads();

    // proto = 0.4*proto + 0.6 * (Z^T @ featH / colsum(Z))  (WMMA, M=5 padded to 16)
    if (tid < WAYS) {
      float cs = 0.0f;
      for (int i = 0; i < NTOT; ++i) cs += Z[i * WAYS + tid];
      colb[tid] = fmaxf(cs, 1e-30f);
    }
    __syncthreads();
    for (int nj = wave; nj < DIM / 16; nj += 8) {
      v8f d0 = {}, d1 = {};
      d0 = wmma16(fragA_Zt(Z,  0), fragB_rm(featH, DIM, nj * 16,  0, NTOT), d0);
      d1 = wmma16(fragA_Zt(Z, 32), fragB_rm(featH, DIM, nj * 16, 32, NTOT), d1);
      d0 = wmma16(fragA_Zt(Z, 64), fragB_rm(featH, DIM, nj * 16, 64, NTOT), d0);
      d1 = wmma16(fragA_Zt(Z, 96), fragB_rm(featH, DIM, nj * 16, 96, NTOT), d1);
      const int n = nj * 16 + (lane & 15);
#pragma unroll
      for (int r = 0; r < 8; ++r) {
        const int row = r + ((lane < 16) ? 0 : 8);             // single M-tile
        if (row < WAYS) {
          const float np = (d0[r] + d1[r]) / colb[row];
          proto[row * DIM + n] = (_Float16)(0.4f * (float)proto[row * DIM + n] + 0.6f * np);
        }
      }
    }
    __syncthreads();
    if (tid < WAYS) {
      float s = 0.0f;
      for (int d = 0; d < DIM; ++d) { const float v = (float)proto[tid * DIM + d]; s += v * v; }
      pnorm[tid] = s;
    }
    __syncthreads();
  }

  // ---- final probabilities, argmax over query rows, accuracy per run ----
  fillZ_wmma(featH, proto, nrm, pnorm, ysl, Z, wave, lane);
  __syncthreads();
  if (wave == 0) otQueryW(Z, lane);
  __syncthreads();
  if (tid == 0) accv[0] = 0.0f;
  __syncthreads();
  if (tid < QS) {
    const float* row = Z + (NS + tid) * WAYS;
    int am = 0; float bv = row[0];
#pragma unroll
    for (int k = 1; k < WAYS; ++k) if (row[k] > bv) { bv = row[k]; am = k; }
    const int y = yqg[run * QS + tid];
    atomicAdd(accv, (am == y) ? 1.0f : 0.0f);
  }
  __syncthreads();
  if (tid == 0) out[run] = accv[0] * (1.0f / (float)QS);
}

extern "C" void kernel_launch(void* const* d_in, const int* in_sizes, int n_in,
                              void* d_out, int out_size, void* d_ws, size_t ws_size,
                              hipStream_t stream) {
  (void)in_sizes; (void)n_in; (void)d_ws; (void)ws_size; (void)out_size;
  const float* xs = (const float*)d_in[0];   // [1000, 25, 640] f32
  const float* xq = (const float*)d_in[1];   // [1000, 75, 640] f32
  const int*   ys = (const int*)d_in[2];     // [1000, 25] i32
  const int*   yq = (const int*)d_in[3];     // [1000, 75] i32
  float*       out = (float*)d_out;          // [1000] f32
  hipLaunchKernelGGL(data_manifolds_kernel, dim3(RUNS), dim3(BLK), 0, stream,
                     xs, xq, ys, yq, out);
}
